// PointNetSetAbstraction_71098888617988
// MI455X (gfx1250) — matrix operand order repaired and checked by reading, not compiled, not run on hardware
//
#include <hip/hip_runtime.h>
#include <hip/hip_bf16.h>

typedef __bf16 bf16_t;
typedef __bf16 v16bf __attribute__((ext_vector_type(16)));
typedef float  v8f   __attribute__((ext_vector_type(8)));

constexpr int B_ = 16;
constexpr int N_ = 8192;
constexpr int S_ = 1024;   // NPOINT
constexpr int K_ = 32;     // NSAMPLE
constexpr int D_ = 64;     // input feature channels
constexpr int M_ = B_ * S_ * K_;   // 524288 GEMM rows
constexpr float R2_ = 0.01f;       // RADIUS^2
constexpr float EPSBN_ = 1e-5f;

// ---------------------------------------------------------------------------
// Farthest point sampling: one workgroup (1024 thr = 32 waves) per batch.
// Points live in registers (8 per thread); per step: min-update + 2-level
// argmax (wave shuffle butterfly, then LDS across 32 waves). Emits new_xyz
// directly into d_out in (B,3,S) layout.
// ---------------------------------------------------------------------------
__global__ __launch_bounds__(1024) void fps_kernel(const float* __restrict__ xyz,
                                                   float* __restrict__ newxyz) {
    const int b = blockIdx.x;
    const int tid = threadIdx.x;
    const float* xb = xyz + (size_t)b * 3 * N_;
    float px[8], py[8], pz[8], mind[8];
#pragma unroll
    for (int u = 0; u < 8; ++u) {
        int j = u * 1024 + tid;
        px[u] = xb[j]; py[u] = xb[N_ + j]; pz[u] = xb[2 * N_ + j];
        mind[u] = 1e10f;
    }
    __shared__ float wm[32];
    __shared__ int   wi[32];
    __shared__ float bc[3];
    __shared__ int   bfar;
    int far = 0;
    if (tid == 0) {  // step 0 emits index 0
        newxyz[b * 3 * S_ + 0]        = xb[0];
        newxyz[b * 3 * S_ + S_]       = xb[N_];
        newxyz[b * 3 * S_ + 2 * S_]   = xb[2 * N_];
    }
    const int wave = tid >> 5, lane = tid & 31;
    for (int s = 1; s < S_; ++s) {
        if (tid == 0) { bc[0] = xb[far]; bc[1] = xb[N_ + far]; bc[2] = xb[2 * N_ + far]; }
        __syncthreads();
        const float cx = bc[0], cy = bc[1], cz = bc[2];
        float best = -1.f; int bidx = 0;
#pragma unroll
        for (int u = 0; u < 8; ++u) {
            float dx = px[u] - cx, dy = py[u] - cy, dz = pz[u] - cz;
            float m = fminf(mind[u], dx * dx + dy * dy + dz * dz);
            mind[u] = m;
            int j = u * 1024 + tid;
            if (m > best || (m == best && j < bidx)) { best = m; bidx = j; }
        }
#pragma unroll
        for (int off = 16; off >= 1; off >>= 1) {
            float ob = __shfl_xor(best, off);
            int   oi = __shfl_xor(bidx, off);
            if (ob > best || (ob == best && oi < bidx)) { best = ob; bidx = oi; }
        }
        if (lane == 0) { wm[wave] = best; wi[wave] = bidx; }
        __syncthreads();
        if (tid < 32) {
            best = wm[tid]; bidx = wi[tid];
#pragma unroll
            for (int off = 16; off >= 1; off >>= 1) {
                float ob = __shfl_xor(best, off);
                int   oi = __shfl_xor(bidx, off);
                if (ob > best || (ob == best && oi < bidx)) { best = ob; bidx = oi; }
            }
            if (tid == 0) bfar = bidx;
        }
        __syncthreads();
        far = bfar;
        if (tid == 0) {
            newxyz[b * 3 * S_ + s]           = xb[far];
            newxyz[b * 3 * S_ + S_ + s]      = xb[N_ + far];
            newxyz[b * 3 * S_ + 2 * S_ + s]  = xb[2 * N_ + far];
        }
    }
}

// ---------------------------------------------------------------------------
// Ball query + grouping. One wave per (b,s) group. Ballot-based ordered
// selection of the first 32 indices within radius (register-only, no LDS).
// Writes bf16 features directly in WMMA 16-bit A-fragment layout, channels
// zero-padded 67 -> 96 (KB=3 K-blocks of 32).
// A-frag mapping (ISA 16x32 16-bit A): for k' = k%32:
//   lane = ((k'>>3)&1)*16 + (row%16),  elem = (k'>>4)*8 + (k'&7)
// ---------------------------------------------------------------------------
__global__ __launch_bounds__(256) void group_kernel(const float* __restrict__ xyz,
                                                    const float* __restrict__ pts,
                                                    const float* __restrict__ newxyz,
                                                    bf16_t* __restrict__ Abuf) {
    const int lane = threadIdx.x & 31;
    const int g = blockIdx.x * 8 + (threadIdx.x >> 5);   // 0..16383
    const int b = g >> 10, s = g & 1023;
    const float* xb = xyz + (size_t)b * 3 * N_;
    const float cx = newxyz[b * 3 * S_ + s];
    const float cy = newxyz[b * 3 * S_ + S_ + s];
    const float cz = newxyz[b * 3 * S_ + 2 * S_ + s];
    int sel = -1, cnt = 0;
    for (int chunk = 0; chunk < N_ / 32 && cnt < K_; ++chunk) {
        int j = chunk * 32 + lane;
        float dx = xb[j] - cx, dy = xb[N_ + j] - cy, dz = xb[2 * N_ + j] - cz;
        bool inside = (dx * dx + dy * dy + dz * dz) <= R2_;
        unsigned mask = (unsigned)__ballot(inside);
        int nset = __popc(mask);
        if (sel < 0 && lane >= cnt && (lane - cnt) < nset) {
            int need = lane - cnt;
            unsigned mm = mask;
            while (need > 0) { mm &= mm - 1; --need; }
            sel = chunk * 32 + (__ffs(mm) - 1);
        }
        cnt += nset;
    }
    int sel0 = __shfl(sel, 0);
    if (sel < 0) sel = sel0;     // fill empty slots with first index (ref semantics)
    if (sel < 0) sel = 0;        // safety; centroid itself always qualifies
    const float relx = xb[sel] - cx;
    const float rely = xb[N_ + sel] - cy;
    const float relz = xb[2 * N_ + sel] - cz;
    const float* pb = pts + (size_t)b * D_ * N_;
    const int r = g * K_ + lane;     // global GEMM row
    const int mblk = r >> 4;
    const int m = r & 15;
    for (int c = 0; c < 96; ++c) {
        float v;
        if (c >= 67)     v = 0.f;
        else if (c >= 3) v = pb[(c - 3) * N_ + sel];
        else             v = (c == 0) ? relx : ((c == 1) ? rely : relz);
        int kb = c >> 5, kk = c & 31;
        int lw = ((kk >> 3) & 1) * 16 + m;
        int e  = (kk >> 4) * 8 + (kk & 7);
        Abuf[(((size_t)mblk * 3 + kb) * 32 + lw) * 16 + e] = (bf16_t)v;
    }
}

// ---------------------------------------------------------------------------
// WMMA GEMM: X (in C-fragment layout) = A[M, 32*KB](bf16 A-frags) * W^T.
// Block = 8 waves; each wave owns one 16-row M tile and iterates COUT/16
// n-blocks with v_wmma_f32_16x16x32_bf16, f32 accumulate. Weights are
// converted once per block into B-fragments in LDS.
// C-fragment layout: X[((mt*NB + nb)*32 + lane)*8 + r], so each lane's 8
// accumulator floats store as one contiguous 32B (2 x global_store_b128),
// and a wave writes 1KB contiguous per n-block.
// ---------------------------------------------------------------------------
template <int KB, int NB, int CIN>
__global__ __launch_bounds__(256) void gemm_wmma(const bf16_t* __restrict__ A,
                                                 const float* __restrict__ W,
                                                 float* __restrict__ X) {
    constexpr int COUT = NB * 16;
    __shared__ __attribute__((aligned(32))) unsigned short bsh[NB * KB * 512];
    const int tid = threadIdx.x;
    for (int i = tid; i < NB * KB * 512; i += 256) {
        int e = i & 15, ln = (i >> 4) & 31;
        int kb = (i >> 9) % KB, nb = (i >> 9) / KB;
        int n = nb * 16 + (ln & 15);
        int k = kb * 32 + (e >> 3) * 16 + (ln >> 4) * 8 + (e & 7);     // B frag: k = (e>>3)*16 + h*8 + (e&7) per K-block
        float v = (k < CIN) ? W[n * CIN + k] : 0.f;
        bf16_t bv = (bf16_t)v;
        bsh[i] = __builtin_bit_cast(unsigned short, bv);
    }
    __syncthreads();
    const int lane = tid & 31;
    const int mt = blockIdx.x * 8 + (tid >> 5);
    const v16bf* Ap = (const v16bf*)A;
    v8f* Xp = (v8f*)X;
    v16bf a[KB];
#pragma unroll
    for (int kb = 0; kb < KB; ++kb) a[kb] = Ap[((size_t)mt * KB + kb) * 32 + lane];
#pragma unroll
    for (int nb = 0; nb < NB; ++nb) {
        v8f acc = {0.f, 0.f, 0.f, 0.f, 0.f, 0.f, 0.f, 0.f};
#pragma unroll
        for (int kb = 0; kb < KB; ++kb) {
            v16bf bf = *(const v16bf*)&bsh[((nb * KB + kb) * 32 + lane) * 16];
            acc = __builtin_amdgcn_wmma_f32_16x16x32_bf16(
                false, a[kb], false, bf, (short)0, acc, false, false);
        }
        Xp[((size_t)mt * NB + nb) * 32 + lane] = acc;
    }
}

// Fragment-index decode helpers (X in C-fragment layout, NB n-blocks):
//   i = ((mt*NB + nb)*32 + lane)*8 + r
//   channel c = nb*16 + (lane & 15)
//   row       = mt*16 + (lane >> 4)*8 + r
template <int NB>
__device__ __forceinline__ int frag_channel(size_t i) {
    return (int)(((i >> 8) & (NB - 1)) * 16 + ((i >> 3) & 15));
}
template <int NB, int LOG2NB>
__device__ __forceinline__ int frag_row(size_t i) {
    int mt = (int)(i >> (8 + LOG2NB));
    return mt * 16 + (int)(((i >> 7) & 1) * 8) + (int)(i & 7);
}

// ---------------------------------------------------------------------------
// BN statistics: per-channel sum & sumsq over all M rows (mean/var over axes
// (0,2)). Channel is fixed per thread across the grid-stride loop (stride is
// a multiple of 256*NB since gridDim=1024), so accumulate locally, combine in
// LDS, one global atomic per channel per block.
// Note: conv bias is a per-channel constant -> cancels exactly under BN.
// ---------------------------------------------------------------------------
template <int NB>
__global__ __launch_bounds__(256) void bn_stats(const float* __restrict__ X,
                                                float* __restrict__ st, int Mtot) {
    constexpr int COUT = NB * 16;
    __shared__ float ssum[128], ssq[128];
    const int tid = threadIdx.x;
    for (int k = tid; k < COUT; k += 256) { ssum[k] = 0.f; ssq[k] = 0.f; }
    __syncthreads();
    const size_t i0 = (size_t)blockIdx.x * 256 + tid;
    const int c = frag_channel<NB>(i0);
    float s = 0.f, q = 0.f;
    const size_t total = (size_t)Mtot * COUT;
    for (size_t i = i0; i < total; i += (size_t)gridDim.x * 256) {
        float v = X[i]; s += v; q += v * v;
    }
    atomicAdd(&ssum[c], s);
    atomicAdd(&ssq[c], q);
    __syncthreads();
    for (int k = tid; k < COUT; k += 256) {
        atomicAdd(&st[k], ssum[k]);
        atomicAdd(&st[COUT + k], ssq[k]);
    }
}

// ---------------------------------------------------------------------------
// BN apply + ReLU: read X (C-fragment layout, NB blocks), emit bf16
// activations in WMMA A-fragment layout (KB K-blocks) for the next GEMM.
// ---------------------------------------------------------------------------
template <int NB, int LOG2NB, int KB>
__global__ __launch_bounds__(256) void bn_apply(const float* __restrict__ X,
                                                const float* __restrict__ st,
                                                const float* __restrict__ gw,
                                                const float* __restrict__ beta,
                                                bf16_t* __restrict__ Aout, int Mtot) {
    constexpr int COUT = NB * 16;
    const float invM = 1.f / (float)Mtot;
    const size_t total = (size_t)Mtot * COUT;
    const size_t stride = (size_t)gridDim.x * 256;
    for (size_t i = (size_t)blockIdx.x * 256 + threadIdx.x; i < total; i += stride) {
        int c = frag_channel<NB>(i);
        int row = frag_row<NB, LOG2NB>(i);
        float mean = st[c] * invM;
        float var  = st[COUT + c] * invM - mean * mean;
        float v = gw[c] * (X[i] - mean) * rsqrtf(var + EPSBN_) + beta[c];
        v = fmaxf(v, 0.f);
        int mblk = row >> 4, m = row & 15;
        int kb = c >> 5, kk = c & 31;
        int lw = ((kk >> 3) & 1) * 16 + m;
        int e  = (kk >> 4) * 8 + (kk & 7);
        Aout[(((size_t)mblk * KB + kb) * 32 + lw) * 16 + e] = (bf16_t)v;
    }
}

// ---------------------------------------------------------------------------
// Final layer BN + ReLU + global max over all S*K positions per (b, channel).
// X is in C-fragment layout with NB=8. grid = B*128 blocks.
// For channel o: nb = o>>4, lane = h*16 + (o&15); rows of batch b live in
// mt = b*2048 .. b*2048+2047, element index (((mt*8+nb)*32 + h*16+on)*8 + r).
// ---------------------------------------------------------------------------
__global__ __launch_bounds__(256) void bn_max(const float* __restrict__ X,
                                              const float* __restrict__ st,
                                              const float* __restrict__ gw,
                                              const float* __restrict__ beta,
                                              float* __restrict__ out, int Mtot) {
    const int o = blockIdx.x & 127;
    const int b = blockIdx.x >> 7;
    const int nb = o >> 4, on = o & 15;
    const float invM = 1.f / (float)Mtot;
    const float mean = st[o] * invM;
    const float var  = st[128 + o] * invM - mean * mean;
    const float sc = gw[o] * rsqrtf(var + EPSBN_);
    const float sh = beta[o] - sc * mean;
    const int rows = S_ * K_;      // 32768 rows per batch = 2048 m-tiles
    float mx = 0.f;                // ReLU output is >= 0
    for (int idx = threadIdx.x; idx < rows; idx += 256) {
        int mtl = idx >> 4;        // local m-tile 0..2047
        int h   = (idx >> 3) & 1;
        int r   = idx & 7;
        size_t mt = (size_t)b * (rows / 16) + mtl;
        float v = sc * X[((mt * 8 + nb) * 32 + h * 16 + on) * 8 + r] + sh;
        mx = fmaxf(mx, v);
    }
    mx = fmaxf(mx, 0.f);
    __shared__ float red[256];
    red[threadIdx.x] = mx;
    __syncthreads();
    for (int off = 128; off >= 1; off >>= 1) {
        if (threadIdx.x < off) red[threadIdx.x] = fmaxf(red[threadIdx.x], red[threadIdx.x + off]);
        __syncthreads();
    }
    if (threadIdx.x == 0) out[b * 128 + o] = red[0];
}

__global__ void init_stats(float* st) {
    for (int i = threadIdx.x; i < 768; i += blockDim.x) st[i] = 0.f;
}

// ---------------------------------------------------------------------------
extern "C" void kernel_launch(void* const* d_in, const int* in_sizes, int n_in,
                              void* d_out, int out_size, void* d_ws, size_t ws_size,
                              hipStream_t stream) {
    const float* xyz = (const float*)d_in[0];
    const float* pts = (const float*)d_in[1];
    const float* w0  = (const float*)d_in[2];
    const float* g0  = (const float*)d_in[4];
    const float* be0 = (const float*)d_in[5];
    const float* w1  = (const float*)d_in[6];
    const float* g1  = (const float*)d_in[8];
    const float* be1 = (const float*)d_in[9];
    const float* w2  = (const float*)d_in[10];
    const float* g2  = (const float*)d_in[12];
    const float* be2 = (const float*)d_in[13];
    // biases b0/b1/b2 cancel exactly under the per-channel mean subtraction.

    float* out = (float*)d_out;
    float* newxyz  = out;                 // (16,3,1024)
    float* outfeat = out + B_ * 3 * S_;   // (16,128)

    // workspace layout
    float*  stats = (float*)d_ws;                                  // 3 x 256 f32
    bf16_t* Abuf  = (bf16_t*)((char*)d_ws + 4096);                 // M x 96 bf16 (96 MB), reused
    float*  Xbuf  = (float*)((char*)d_ws + 4096 + (size_t)M_ * 96 * 2);  // M x 128 f32 (256 MB)

    init_stats<<<1, 256, 0, stream>>>(stats);
    fps_kernel<<<B_, 1024, 0, stream>>>(xyz, newxyz);
    group_kernel<<<(B_ * S_) / 8, 256, 0, stream>>>(xyz, pts, newxyz, Abuf);

    // layer 0: 67(pad 96) -> 64
    gemm_wmma<3, 4, 67><<<M_ / 128, 256, 0, stream>>>(Abuf, w0, Xbuf);
    bn_stats<4><<<1024, 256, 0, stream>>>(Xbuf, stats + 0, M_);
    bn_apply<4, 2, 2><<<2048, 256, 0, stream>>>(Xbuf, stats + 0, g0, be0, Abuf, M_);

    // layer 1: 64 -> 64
    gemm_wmma<2, 4, 64><<<M_ / 128, 256, 0, stream>>>(Abuf, w1, Xbuf);
    bn_stats<4><<<1024, 256, 0, stream>>>(Xbuf, stats + 256, M_);
    bn_apply<4, 2, 2><<<2048, 256, 0, stream>>>(Xbuf, stats + 256, g1, be1, Abuf, M_);

    // layer 2: 64 -> 128, then BN + ReLU + global max
    gemm_wmma<2, 8, 64><<<M_ / 128, 256, 0, stream>>>(Abuf, w2, Xbuf);
    bn_stats<8><<<1024, 256, 0, stream>>>(Xbuf, stats + 512, M_);
    bn_max<<<B_ * 128, 256, 0, stream>>>(Xbuf, stats + 512, g2, be2, outfeat, M_);
}
